// GCNv3_23862838296799
// MI455X (gfx1250) — matrix operand order repaired and checked by reading, not compiled, not run on hardware
//
#include <hip/hip_runtime.h>
#include <hip/hip_bf16.h>

#define N_NODES    100000
#define N_EDGES    800000
#define HID        128
#define NUM_GRAPHS 64
#define ROWS_PER_BLOCK 64

typedef __attribute__((ext_vector_type(16))) __bf16 v16bf;
typedef __attribute__((ext_vector_type(8)))  float  v8f;

__device__ __forceinline__ float lrelu(float v) { return v > 0.f ? v : 0.01f * v; }

// ---------------- f32 -> bf16 weight conversion WITH fragment swizzle ----------------
// Output layout: o[(((t*4 + c)*32 + lane)*16 + e]  where
//   t = column tile (0..7), c = K-chunk (0..3), lane = wave lane (0..31),
//   e = 2p+j = element within the 16-bf16 B fragment.
// Maps to W[k][n] with n = t*16 + (lane&15), k = 32c + 16*(lane>>4) + e.
// A wave's B fragment for (tile t, chunk c) is then 16 consecutive bf16 per lane,
// 32-byte aligned -> loads as 2x ds_load_b128.
__global__ __launch_bounds__(256)
void k_cvt_swz(const float* __restrict__ w, __bf16* __restrict__ o) {
  int i = blockIdx.x * 256 + threadIdx.x;   // 0..16383
  if (i >= 16384) return;
  int e = i & 15;
  int l = (i >> 4) & 31;
  int c = (i >> 9) & 3;
  int t = i >> 11;
  int n = t * 16 + (l & 15);
  int k = 32 * c + 16 * (l >> 4) + e;
  o[i] = (__bf16)w[k * 128 + n];
}

// ---------------- zero fill ----------------
__global__ __launch_bounds__(256)
void k_zero(float* __restrict__ p, int n) {
  int i = blockIdx.x * 256 + threadIdx.x;
  if (i < n) p[i] = 0.f;
}

// ---------------- out = lrelu(A1@W1 (+ A2@W2) + b), [*,128]@[128,128] ----------------
// Block: 256 threads (8 wave32), covers 64 rows x 128 cols.
// Wave = one 16-col tile; holds 4 B fragments (full K) in regs, sweeps 4 M-tiles.
__global__ __launch_bounds__(256)
void k_gemm128(const float* __restrict__ A1, const __bf16* __restrict__ W1sw,
               const float* __restrict__ A2, const __bf16* __restrict__ W2sw,
               const float* __restrict__ bias, float* __restrict__ out, int nrows)
{
  __shared__ __align__(32) __bf16 aLds[ROWS_PER_BLOCK * 128];  // 16 KB
  __shared__ __align__(32) __bf16 wLds[128 * 128];             // 32 KB (swizzled)

  const int tid  = threadIdx.x;
  const int wave = tid >> 5;
  const int lane = tid & 31;
  const int row0 = blockIdx.x * ROWS_PER_BLOCK;
  const int m    = lane & 15;    // A row in tile / output column in tile
  const int half = lane >> 4;
  const int n0   = wave * 16;

  const int nsrc = (A2 != nullptr) ? 2 : 1;
  v8f acc[4] = {v8f{}, v8f{}, v8f{}, v8f{}};

  for (int s = 0; s < nsrc; ++s) {
    const float*  A = s ? A2 : A1;
    const __bf16* W = s ? W2sw : W1sw;

    __syncthreads();  // protect LDS reuse across source phases
    // Stage activation tile (convert f32 -> bf16), 64x128
    for (int i = tid; i < ROWS_PER_BLOCK * 128; i += 256) {
      int r = i >> 7, k = i & 127;
      int row = row0 + r;
      aLds[i] = (row < nrows) ? (__bf16)A[(size_t)row * 128 + k] : (__bf16)0.f;
    }
    // Stage swizzled weights: straight dword copy
    {
      const uint32_t* Wu = (const uint32_t*)W;
      uint32_t*       Lu = (uint32_t*)wLds;
      for (int i = tid; i < (128 * 128) / 2; i += 256) Lu[i] = Wu[i];
    }
    __syncthreads();

    // This wave's 4 B fragments (one per K-chunk), contiguous & aligned in LDS
    v16bf bfrag[4];
#pragma unroll
    for (int c = 0; c < 4; ++c)
      bfrag[c] = *(const v16bf*)&wLds[(((wave * 4 + c) * 32) + lane) * 16];

#pragma unroll
    for (int mt = 0; mt < 4; ++mt) {
      const __bf16* aBase = &aLds[(mt * 16 + m) * 128];
#pragma unroll
      for (int c = 0; c < 4; ++c) {
        // A 16x32 bf16 fragment: 16 contiguous k per lane half (k = 16*half + e), matching B swizzle
        v16bf a;
#pragma unroll
        for (int e = 0; e < 8; ++e) {
          a[e]     = aBase[32 * c + half * 16 + e];
          a[8 + e] = aBase[32 * c + half * 16 + 8 + e];
        }
        acc[mt] = __builtin_amdgcn_wmma_f32_16x16x32_bf16(
            false, a, false, bfrag[c], (short)0, acc[mt], false, false);
      }
    }
  }

  const float bv = bias[n0 + m];
#pragma unroll
  for (int mt = 0; mt < 4; ++mt) {
#pragma unroll
    for (int r = 0; r < 8; ++r) {
      // C/D layout: VGPR r -> row (r + 8*half), lane%16 -> column
      int row = row0 + mt * 16 + r + 8 * half;
      if (row < nrows) out[(size_t)row * 128 + n0 + m] = lrelu(acc[mt][r] + bv);
    }
  }
}

// ---------------- edge gather + scatter-add: agg[dst] += h[src] ----------------
__global__ __launch_bounds__(256)
void k_scatter(const float* __restrict__ h, const int* __restrict__ src,
               const int* __restrict__ dst, float* __restrict__ agg, int nEdges)
{
  int t = blockIdx.x * 256 + threadIdx.x;
  int e = t >> 5;
  int lane = t & 31;
  if (e >= nEdges) return;
  int s = src[e], d = dst[e];
  float4 v = ((const float4*)(h + (size_t)s * 128))[lane];
  float* ap = agg + (size_t)d * 128 + lane * 4;
  atomicAdd(ap + 0, v.x);
  atomicAdd(ap + 1, v.y);
  atomicAdd(ap + 2, v.z);
  atomicAdd(ap + 3, v.w);
}

// ---------------- segment mean-pool prep (batch sorted: run-length accumulate) ----------------
#define POOL_NODES 512
__global__ __launch_bounds__(128)
void k_pool(const float* __restrict__ h, const int* __restrict__ batch,
            float* __restrict__ sums, float* __restrict__ cnt, int nNodes)
{
  int f     = threadIdx.x;
  int start = blockIdx.x * POOL_NODES;
  int end   = start + POOL_NODES;
  if (end > nNodes) end = nNodes;
  if (start >= end) return;

  int   curg = batch[start];
  float run  = 0.f;
  int   runc = 0;
  for (int n = start; n < end; ++n) {
    int g = batch[n];
    if (g != curg) {
      atomicAdd(&sums[curg * 128 + f], run);
      if (f == 0) atomicAdd(&cnt[curg], (float)runc);
      run = 0.f; runc = 0; curg = g;
    }
    run += h[(size_t)n * 128 + f];
    runc++;
  }
  atomicAdd(&sums[curg * 128 + f], run);
  if (f == 0) atomicAdd(&cnt[curg], (float)runc);
}

// ---------------- head: g = sums/cnt; fc3; fc4; fc5 -> out[64,2] ----------------
__global__ __launch_bounds__(128)
void k_head(const float* __restrict__ sums, const float* __restrict__ cnt,
            const float* __restrict__ w3, const float* __restrict__ b3,
            const float* __restrict__ w4, const float* __restrict__ b4,
            const float* __restrict__ w5, const float* __restrict__ b5,
            float* __restrict__ out)
{
  __shared__ float t3[NUM_GRAPHS][128];
  __shared__ float t4[NUM_GRAPHS][64];
  __shared__ float inv[NUM_GRAPHS];
  int f = threadIdx.x;
  if (f < NUM_GRAPHS) inv[f] = 1.f / fmaxf(cnt[f], 1.f);
  __syncthreads();

  for (int gi = 0; gi < NUM_GRAPHS; ++gi) {
    float acc = b3[f];
    float iv  = inv[gi];
    for (int k = 0; k < 128; ++k)
      acc += (sums[gi * 128 + k] * iv) * w3[k * 128 + f];
    t3[gi][f] = lrelu(acc);
  }
  __syncthreads();

  if (f < 64) {
    for (int gi = 0; gi < NUM_GRAPHS; ++gi) {
      float acc = b4[f];
      for (int k = 0; k < 128; ++k) acc += t3[gi][k] * w4[k * 64 + f];
      t4[gi][f] = lrelu(acc);
    }
  }
  __syncthreads();

  int gi = f >> 1, cls = f & 1;
  float acc = b5[cls];
  for (int k = 0; k < 64; ++k) acc += t4[gi][k] * w5[k * 2 + cls];
  out[gi * 2 + cls] = acc;
}

// ---------------- orchestration ----------------
extern "C" void kernel_launch(void* const* d_in, const int* in_sizes, int n_in,
                              void* d_out, int out_size, void* d_ws, size_t ws_size,
                              hipStream_t stream)
{
  (void)in_sizes; (void)n_in; (void)out_size; (void)ws_size;

  const float* x     = (const float*)d_in[0];
  const int*   ei    = (const int*)d_in[1];
  const int*   batch = (const int*)d_in[2];
  const float* fc1_w = (const float*)d_in[3];
  const float* fc1_b = (const float*)d_in[4];
  const float* c1rw  = (const float*)d_in[5];
  const float* c1rb  = (const float*)d_in[6];
  const float* c1tw  = (const float*)d_in[7];
  const float* fc2_w = (const float*)d_in[8];
  const float* fc2_b = (const float*)d_in[9];
  const float* c2rw  = (const float*)d_in[10];
  const float* c2rb  = (const float*)d_in[11];
  const float* c2tw  = (const float*)d_in[12];
  const float* fc3_w = (const float*)d_in[13];
  const float* fc3_b = (const float*)d_in[14];
  const float* fc4_w = (const float*)d_in[15];
  const float* fc4_b = (const float*)d_in[16];
  const float* fc5_w = (const float*)d_in[17];
  const float* fc5_b = (const float*)d_in[18];
  const int* src = ei;
  const int* dst = ei + N_EDGES;

  const size_t NM = (size_t)N_NODES * HID;
  char* ws    = (char*)d_ws;
  float* buf0 = (float*)ws;
  float* buf1 = (float*)(ws + NM * 4);
  float* buf2 = (float*)(ws + 2 * NM * 4);
  float* sums = (float*)(ws + 3 * NM * 4);
  float* cnt  = sums + NUM_GRAPHS * HID;
  __bf16* wb  = (__bf16*)(cnt + NUM_GRAPHS);
  __bf16* w_fc1 = wb;
  __bf16* w_c1r = wb + 1 * 16384;
  __bf16* w_c1t = wb + 2 * 16384;
  __bf16* w_fc2 = wb + 3 * 16384;
  __bf16* w_c2r = wb + 4 * 16384;
  __bf16* w_c2t = wb + 5 * 16384;

  // weight conversions (f32 -> bf16, fragment-swizzled), 128x128 each
  k_cvt_swz<<<64, 256, 0, stream>>>(fc1_w, w_fc1);
  k_cvt_swz<<<64, 256, 0, stream>>>(c1rw,  w_c1r);
  k_cvt_swz<<<64, 256, 0, stream>>>(c1tw,  w_c1t);
  k_cvt_swz<<<64, 256, 0, stream>>>(fc2_w, w_fc2);
  k_cvt_swz<<<64, 256, 0, stream>>>(c2rw,  w_c2r);
  k_cvt_swz<<<64, 256, 0, stream>>>(c2tw,  w_c2t);

  const int gemmGrid = (N_NODES + ROWS_PER_BLOCK - 1) / ROWS_PER_BLOCK;  // 1563
  const int zeroGrid = (int)((NM + 255) / 256);
  const int scatGrid = (N_EDGES * 32 + 255) / 256;
  const int poolGrid = (N_NODES + POOL_NODES - 1) / POOL_NODES;

  // h1 = lrelu(x @ fc1_w + fc1_b)
  k_gemm128<<<gemmGrid, 256, 0, stream>>>(x, w_fc1, nullptr, nullptr, fc1_b, buf0, N_NODES);

  // conv1
  k_zero<<<zeroGrid, 256, 0, stream>>>(buf1, (int)NM);
  k_scatter<<<scatGrid, 256, 0, stream>>>(buf0, src, dst, buf1, N_EDGES);
  k_gemm128<<<gemmGrid, 256, 0, stream>>>(buf1, w_c1r, buf0, w_c1t, c1rb, buf2, N_NODES);

  // h3 = lrelu(h2 @ fc2_w + fc2_b)
  k_gemm128<<<gemmGrid, 256, 0, stream>>>(buf2, w_fc2, nullptr, nullptr, fc2_b, buf0, N_NODES);

  // conv2
  k_zero<<<zeroGrid, 256, 0, stream>>>(buf1, (int)NM);
  k_scatter<<<scatGrid, 256, 0, stream>>>(buf0, src, dst, buf1, N_EDGES);
  k_gemm128<<<gemmGrid, 256, 0, stream>>>(buf1, w_c2r, buf0, w_c2t, c2rb, buf2, N_NODES);

  // mean pool + head
  k_zero<<<((NUM_GRAPHS * HID + NUM_GRAPHS) + 255) / 256, 256, 0, stream>>>(
      sums, NUM_GRAPHS * HID + NUM_GRAPHS);
  k_pool<<<poolGrid, 128, 0, stream>>>(buf2, batch, sums, cnt, N_NODES);
  k_head<<<1, 128, 0, stream>>>(sums, cnt, fc3_w, fc3_b, fc4_w, fc4_b, fc5_w, fc5_b,
                                (float*)d_out);
}
